// Scaled_Dot_Product_Attention_52407190946412
// MI455X (gfx1250) — compile-verified, matrix operand order checked by reading, not run on hardware
//
#include <hip/hip_runtime.h>
#include <hip/hip_bf16.h>
#include <math.h>

typedef __attribute__((ext_vector_type(16))) _Float16 v16h;
typedef __attribute__((ext_vector_type(8)))  _Float16 v8h;
typedef __attribute__((ext_vector_type(8)))  float    v8f;
typedef __attribute__((vector_size(16)))     int      v4i_vec;

#define ATT_B 16
#define ATT_N 4096
#define ATT_D 64
#define LOG2E 1.44269504088896340736f

// ---- CDNA5 async-to-LDS availability --------------------------------------
#if defined(__has_builtin)
#  if __has_builtin(__builtin_amdgcn_global_load_async_to_lds_b128)
#    define ATT_HAVE_ASYNC 1
#  endif
#endif
#ifndef ATT_HAVE_ASYNC
#  define ATT_HAVE_ASYNC 0
#endif

__device__ __forceinline__ float exp2_raw(float x) {
#if defined(__has_builtin) && __has_builtin(__builtin_amdgcn_exp2f)
    return __builtin_amdgcn_exp2f(x);   // raw v_exp_f32, no range-guard code
#else
    return exp2f(x);
#endif
}

__device__ __forceinline__ v8f wmma_f16(v16h a, v16h b, v8f c) {
    return __builtin_amdgcn_wmma_f32_16x16x32_f16(false, a, false, b, (short)0, c, false, false);
}

// Two contiguous 8-half (16B) loads -> 16-half operand matching the CDNA5
// 16-bit A/B 16x32 layout: lane<16 K={0..7,16..23}, lane>=16 K={8..15,24..31}.
__device__ __forceinline__ v16h load_pair(const _Float16* p0, const _Float16* p1) {
    v8h a = *(const v8h*)p0;
    v8h b = *(const v8h*)p1;
    return __builtin_shufflevector(a, b, 0, 1, 2, 3, 4, 5, 6, 7,
                                         8, 9, 10, 11, 12, 13, 14, 15);
}

// Stage 16B global -> LDS. Async (ASYNCcnt-tracked) when available.
__device__ __forceinline__ void stage_b128(const _Float16* g, _Float16* l) {
#if ATT_HAVE_ASYNC
    __builtin_amdgcn_global_load_async_to_lds_b128(
        (__attribute__((address_space(1))) v4i_vec*)g,
        (__attribute__((address_space(3))) v4i_vec*)l, 0, 0);
#else
    *(v8h*)l = *(const v8h*)g;
#endif
}

__device__ __forceinline__ void wait_async_prev() {   // own prev tile done (<=8 outstanding)
#if ATT_HAVE_ASYNC
#if defined(__has_builtin) && __has_builtin(__builtin_amdgcn_s_wait_asynccnt)
    __builtin_amdgcn_s_wait_asynccnt(8);
#else
    asm volatile("s_wait_asynccnt 0x8" ::: "memory");
#endif
#endif
}

__device__ __forceinline__ void wait_async_all() {
#if ATT_HAVE_ASYNC
#if defined(__has_builtin) && __has_builtin(__builtin_amdgcn_s_wait_asynccnt)
    __builtin_amdgcn_s_wait_asynccnt(0);
#else
    asm volatile("s_wait_asynccnt 0x0" ::: "memory");
#endif
#endif
}

// ---------------------------------------------------------------------------
// Prep kernel 1: elementwise f32 -> f16 with scale (Q gets LOG2E folded in).
// ---------------------------------------------------------------------------
__global__ __launch_bounds__(256)
void cvt_f32_to_f16_kernel(const float* __restrict__ src, _Float16* __restrict__ dst,
                           float scale) {
    const size_t i = ((size_t)blockIdx.x * 256 + threadIdx.x) * 8;
    const float4 a = *(const float4*)(src + i);
    const float4 b = *(const float4*)(src + i + 4);
    v8h r;
    r[0] = (_Float16)(a.x * scale); r[1] = (_Float16)(a.y * scale);
    r[2] = (_Float16)(a.z * scale); r[3] = (_Float16)(a.w * scale);
    r[4] = (_Float16)(b.x * scale); r[5] = (_Float16)(b.y * scale);
    r[6] = (_Float16)(b.z * scale); r[7] = (_Float16)(b.w * scale);
    *(v8h*)(dst + i) = r;
}

// ---------------------------------------------------------------------------
// Prep kernel 2: V [B,N,64] f32 -> Vt [B,64,N] f16 (LDS-tiled transpose).
// ---------------------------------------------------------------------------
__global__ __launch_bounds__(256)
void transpose_cvt_kernel(const float* __restrict__ V, _Float16* __restrict__ Vt) {
    __shared__ _Float16 tile[64][ATT_D + 4];
    const int b  = blockIdx.y;
    const int n0 = blockIdx.x * 64;
    const float* __restrict__ Vb  = V  + ((size_t)b * ATT_N + n0) * ATT_D;
    _Float16* __restrict__     Vtb = Vt + (size_t)b * ATT_D * ATT_N;
    {
        const int r  = threadIdx.x >> 2;
        const int dg = (threadIdx.x & 3) * 16;
        const float* src = Vb + (size_t)r * ATT_D + dg;
#pragma unroll
        for (int i = 0; i < 16; i += 4) {
            const float4 f = *(const float4*)(src + i);
            tile[r][dg + i + 0] = (_Float16)f.x;
            tile[r][dg + i + 1] = (_Float16)f.y;
            tile[r][dg + i + 2] = (_Float16)f.z;
            tile[r][dg + i + 3] = (_Float16)f.w;
        }
    }
    __syncthreads();
    {
        const int d  = threadIdx.x >> 2;
        const int ng = (threadIdx.x & 3) * 16;
        v8h p0, p1;
#pragma unroll
        for (int i = 0; i < 8; ++i) p0[i] = tile[ng + i][d];
#pragma unroll
        for (int i = 0; i < 8; ++i) p1[i] = tile[ng + 8 + i][d];
        _Float16* dst = Vtb + (size_t)d * ATT_N + n0 + ng;
        *(v8h*)(dst)     = p0;
        *(v8h*)(dst + 8) = p1;
    }
}

// ---------------------------------------------------------------------------
// Main kernel: 4 waves / workgroup, each wave owns a 16-query tile.
// 64-key iterations; K and V^T tiles shared via LDS, double-buffered with
// async-to-LDS loads. S^T / O^T formulation keeps softmax state per-lane.
// ---------------------------------------------------------------------------
__global__ __launch_bounds__(128)
void attn_wmma_f16_kernel(const _Float16* __restrict__ Qh,
                          const _Float16* __restrict__ Kh,
                          const _Float16* __restrict__ Vth,
                          float* __restrict__ O)
{
    __shared__ _Float16 kbuf[2][64 * 64];   // [buf][key][d]   (128B rows)
    __shared__ _Float16 vbuf[2][64 * 64];   // [buf][d][key]   (128B rows)

    const int tid  = threadIdx.x;
    const int lane = tid & 31;
    const int w    = tid >> 5;          // wave id 0..3
    const int lo   = lane & 15;
    const int h    = lane >> 4;
    const int hb   = h * 8;

    const int b  = blockIdx.y;
    const int q0 = blockIdx.x * 64 + w * 16;

    const _Float16* __restrict__ Qb = Qh  + (size_t)b * ATT_N * ATT_D;
    const _Float16* __restrict__ Kb = Kh  + (size_t)b * ATT_N * ATT_D;
    const _Float16* __restrict__ Vb = Vth + (size_t)b * ATT_D * ATT_N;
    float* __restrict__           Ob = O  + (size_t)b * ATT_N * ATT_D;

    // Staging assignment: wave w covers rows w*16 + j*4 + (lane>>3), 16B cols.
    const int sr = (lane >> 3);         // 0..3
    const int sc = (lane & 7) * 8;      // halves

    // Q tile as B operand, pre-scaled by LOG2E in prep.
    const _Float16* qrow = Qb + (size_t)(q0 + lo) * ATT_D;
    const v16h qb0 = load_pair(qrow + hb,      qrow + 16 + hb);
    const v16h qb1 = load_pair(qrow + 32 + hb, qrow + 48 + hb);

    v8f o0 = {}, o1 = {}, o2 = {}, o3 = {};
    float m = -__builtin_inff();
    float l = 0.0f;

    // Prologue: stage tile 0 into buffer 0.
    {
        const _Float16* gk = Kb + (size_t)(w * 16 + sr) * ATT_D + sc;
        _Float16*       lk = &kbuf[0][(w * 16 + sr) * 64 + sc];
        const _Float16* gv = Vb + (size_t)(w * 16 + sr) * ATT_N + sc;
        _Float16*       lv = &vbuf[0][(w * 16 + sr) * 64 + sc];
#pragma unroll
        for (int j = 0; j < 4; ++j) {
            stage_b128(gk + j * 4 * ATT_D, lk + j * 256);
            stage_b128(gv + (size_t)j * 4 * ATT_N, lv + j * 256);
        }
    }

    int bb = 0;
    for (int kt = 0; kt < ATT_N; kt += 64) {
        __syncthreads();   // all waves done reading buf bb^1 (prev iteration)

        if (kt + 64 < ATT_N) {   // stage next tile into the other buffer
            const int nb = bb ^ 1;
            const _Float16* gk = Kb + (size_t)(kt + 64 + w * 16 + sr) * ATT_D + sc;
            _Float16*       lk = &kbuf[nb][(w * 16 + sr) * 64 + sc];
            const _Float16* gv = Vb + (size_t)(w * 16 + sr) * ATT_N + (kt + 64) + sc;
            _Float16*       lv = &vbuf[nb][(w * 16 + sr) * 64 + sc];
#pragma unroll
            for (int j = 0; j < 4; ++j) {
                stage_b128(gk + j * 4 * ATT_D, lk + j * 256);
                stage_b128(gv + (size_t)j * 4 * ATT_N, lv + j * 256);
            }
            wait_async_prev();   // this wave's current-tile (older) ops done
        } else {
            wait_async_all();
        }
        __syncthreads();   // everyone's current-tile data visible in LDS

        const _Float16* kb = kbuf[bb];
        const _Float16* vb = vbuf[bb];
        const v8f zero = {};

        // ---- S^T: 4 key-group tiles of 16x16, d-dim in two 32-chunks.
        v8f st0, st1, st2, st3;
#define ATT_ST_TILE(stv, kg)                                          \
        {                                                             \
            const _Float16* kr = kb + ((kg) * 16 + lo) * 64;          \
            v16h a0 = load_pair(kr + hb,      kr + 16 + hb);          \
            v16h a1 = load_pair(kr + 32 + hb, kr + 48 + hb);          \
            stv = wmma_f16(a0, qb0, zero);                            \
            stv = wmma_f16(a1, qb1, stv);                             \
        }
        ATT_ST_TILE(st0, 0)
        ATT_ST_TILE(st1, 1)
        ATT_ST_TILE(st2, 2)
        ATT_ST_TILE(st3, 3)
#undef ATT_ST_TILE

        // ---- online softmax (log2 domain; partner lane = xor 16).
        float tmax = st0[0];
#pragma unroll
        for (int j = 1; j < 8; ++j) tmax = fmaxf(tmax, st0[j]);
#pragma unroll
        for (int j = 0; j < 8; ++j) tmax = fmaxf(tmax, st1[j]);
#pragma unroll
        for (int j = 0; j < 8; ++j) tmax = fmaxf(tmax, st2[j]);
#pragma unroll
        for (int j = 0; j < 8; ++j) tmax = fmaxf(tmax, st3[j]);
        tmax = fmaxf(tmax, __shfl_xor(tmax, 16, 32));

        const float mnew = fmaxf(m, tmax);
        if (__ballot(mnew > m)) {            // wave-uniform: rescale only on max update
            const float alpha = exp2_raw(m - mnew);   // ==1 for lanes without update
            l *= alpha;
#pragma unroll
            for (int j = 0; j < 8; ++j) {
                o0[j] *= alpha; o1[j] *= alpha; o2[j] *= alpha; o3[j] *= alpha;
            }
        }
        m = mnew;

        // ---- P^T operands: per-lane exp2 of the S^T C-layout tiles.
        v16h pb0, pb1;
        float psum = 0.0f;
#pragma unroll
        for (int j = 0; j < 8; ++j) {
            const float p = exp2_raw(st0[j] - mnew);
            psum += p; pb0[j] = (_Float16)p;
        }
#pragma unroll
        for (int j = 0; j < 8; ++j) {
            const float p = exp2_raw(st1[j] - mnew);
            psum += p; pb0[8 + j] = (_Float16)p;
        }
#pragma unroll
        for (int j = 0; j < 8; ++j) {
            const float p = exp2_raw(st2[j] - mnew);
            psum += p; pb1[j] = (_Float16)p;
        }
#pragma unroll
        for (int j = 0; j < 8; ++j) {
            const float p = exp2_raw(st3[j] - mnew);
            psum += p; pb1[8 + j] = (_Float16)p;
        }
        psum += __shfl_xor(psum, 16, 32);
        l += psum;

        // ---- O^T += V^T x P^T: 4 d-groups x 2 key chunks.
#define ATT_OT_ACC(og, g)                                             \
        {                                                             \
            const _Float16* vr = vb + ((g) * 16 + lo) * 64;           \
            v16h va0 = load_pair(vr + hb,      vr + 16 + hb);         \
            v16h va1 = load_pair(vr + 32 + hb, vr + 48 + hb);         \
            og = wmma_f16(va0, pb0, og);                              \
            og = wmma_f16(va1, pb1, og);                              \
        }
        ATT_OT_ACC(o0, 0)
        ATT_OT_ACC(o1, 1)
        ATT_OT_ACC(o2, 2)
        ATT_OT_ACC(o3, 3)
#undef ATT_OT_ACC

        bb ^= 1;
    }

    // ---- normalize + store: tile g, vgpr j holds d = g*16 + 8h + j, col = lo.
    const float inv = 1.0f / l;
    float* orow = Ob + (size_t)(q0 + lo) * ATT_D + hb;
#define ATT_STORE_G(og, g)                                                           \
    {                                                                                \
        float4 s0 = make_float4(og[0] * inv, og[1] * inv, og[2] * inv, og[3] * inv); \
        float4 s1 = make_float4(og[4] * inv, og[5] * inv, og[6] * inv, og[7] * inv); \
        *(float4*)(orow + (g) * 16)     = s0;                                        \
        *(float4*)(orow + (g) * 16 + 4) = s1;                                        \
    }
    ATT_STORE_G(o0, 0)
    ATT_STORE_G(o1, 1)
    ATT_STORE_G(o2, 2)
    ATT_STORE_G(o3, 3)
#undef ATT_STORE_G
}

// ---------------------------------------------------------------------------
// Fallback: direct-f32 version (used only if workspace is too small).
// ---------------------------------------------------------------------------
__device__ __forceinline__ v16h load_row_f16(const float* __restrict__ rowp, int c, int h) {
    const int d0 = c * 32 + h * 8;
    const float4 f0 = *(const float4*)(rowp + d0);
    const float4 f1 = *(const float4*)(rowp + d0 + 4);
    const float4 g0 = *(const float4*)(rowp + d0 + 16);
    const float4 g1 = *(const float4*)(rowp + d0 + 20);
    v16h r;
    r[0]  = (_Float16)f0.x; r[1]  = (_Float16)f0.y; r[2]  = (_Float16)f0.z; r[3]  = (_Float16)f0.w;
    r[4]  = (_Float16)f1.x; r[5]  = (_Float16)f1.y; r[6]  = (_Float16)f1.z; r[7]  = (_Float16)f1.w;
    r[8]  = (_Float16)g0.x; r[9]  = (_Float16)g0.y; r[10] = (_Float16)g0.z; r[11] = (_Float16)g0.w;
    r[12] = (_Float16)g1.x; r[13] = (_Float16)g1.y; r[14] = (_Float16)g1.z; r[15] = (_Float16)g1.w;
    return r;
}

__device__ __forceinline__ v16h load_vt_f16(const float* __restrict__ Vb,
                                            int kbase0, int kbase1, int dcol) {
    v16h r;
#pragma unroll
    for (int i = 0; i < 8; ++i) r[i]     = (_Float16)Vb[(size_t)(kbase0 + i) * ATT_D + dcol];
#pragma unroll
    for (int i = 0; i < 8; ++i) r[8 + i] = (_Float16)Vb[(size_t)(kbase1 + i) * ATT_D + dcol];
    return r;
}

__global__ __launch_bounds__(32)
void attn_wmma_f32src_kernel(const float* __restrict__ Q, const float* __restrict__ K,
                             const float* __restrict__ V, float* __restrict__ O)
{
    const int lane = threadIdx.x & 31;
    const int lo   = lane & 15;
    const int h    = lane >> 4;
    const int b  = blockIdx.y;
    const int q0 = blockIdx.x * 16;

    const size_t boff = (size_t)b * ATT_N * ATT_D;
    const float* __restrict__ Qb = Q + boff;
    const float* __restrict__ Kb = K + boff;
    const float* __restrict__ Vb = V + boff;
    float* __restrict__       Ob = O + boff;

    const float* qrow = Qb + (size_t)(q0 + lo) * ATT_D;
    const v16h qb0 = load_row_f16(qrow, 0, h);
    const v16h qb1 = load_row_f16(qrow, 1, h);

    v8f o0 = {}, o1 = {}, o2 = {}, o3 = {};
    float m = -__builtin_inff();
    float l = 0.0f;

    for (int kt = 0; kt < ATT_N; kt += 32) {
        const float* krow0 = Kb + (size_t)(kt + lo) * ATT_D;
        const float* krow1 = Kb + (size_t)(kt + 16 + lo) * ATT_D;
        const v8f zero = {};

        v16h ka0 = load_row_f16(krow0, 0, h);
        v16h ka1 = load_row_f16(krow0, 1, h);
        v8f st0 = wmma_f16(ka0, qb0, zero);
        st0 = wmma_f16(ka1, qb1, st0);

        v16h kb0c = load_row_f16(krow1, 0, h);
        v16h kb1c = load_row_f16(krow1, 1, h);
        v8f st1 = wmma_f16(kb0c, qb0, zero);
        st1 = wmma_f16(kb1c, qb1, st1);

        float tmax = st0[0];
#pragma unroll
        for (int j = 1; j < 8; ++j) tmax = fmaxf(tmax, st0[j]);
#pragma unroll
        for (int j = 0; j < 8; ++j) tmax = fmaxf(tmax, st1[j]);
        tmax = fmaxf(tmax, __shfl_xor(tmax, 16, 32));

        const float mnew  = fmaxf(m, tmax);
        const float alpha = exp2_raw((m - mnew) * LOG2E);

        v16h pb;
        float psum = 0.0f;
#pragma unroll
        for (int j = 0; j < 8; ++j) {
            const float p = exp2_raw((st0[j] - mnew) * LOG2E);
            psum += p; pb[j] = (_Float16)p;
        }
#pragma unroll
        for (int j = 0; j < 8; ++j) {
            const float p = exp2_raw((st1[j] - mnew) * LOG2E);
            psum += p; pb[8 + j] = (_Float16)p;
        }
        psum += __shfl_xor(psum, 16, 32);
        l = l * alpha + psum;
        m = mnew;

#pragma unroll
        for (int j = 0; j < 8; ++j) { o0[j] *= alpha; o1[j] *= alpha; o2[j] *= alpha; o3[j] *= alpha; }

        const int kbase0 = kt + h * 8;
        const int kbase1 = kt + 16 + h * 8;
        v16h va;
        va = load_vt_f16(Vb, kbase0, kbase1,  0 + lo); o0 = wmma_f16(va, pb, o0);
        va = load_vt_f16(Vb, kbase0, kbase1, 16 + lo); o1 = wmma_f16(va, pb, o1);
        va = load_vt_f16(Vb, kbase0, kbase1, 32 + lo); o2 = wmma_f16(va, pb, o2);
        va = load_vt_f16(Vb, kbase0, kbase1, 48 + lo); o3 = wmma_f16(va, pb, o3);
    }

    const float inv = 1.0f / l;
    float* orow = Ob + (size_t)(q0 + lo) * ATT_D + h * 8;
#define ATT_STORE_G(og, g)                                                           \
    {                                                                                \
        float4 s0 = make_float4(og[0] * inv, og[1] * inv, og[2] * inv, og[3] * inv); \
        float4 s1 = make_float4(og[4] * inv, og[5] * inv, og[6] * inv, og[7] * inv); \
        *(float4*)(orow + (g) * 16)     = s0;                                        \
        *(float4*)(orow + (g) * 16 + 4) = s1;                                        \
    }
    ATT_STORE_G(o0, 0)
    ATT_STORE_G(o1, 1)
    ATT_STORE_G(o2, 2)
    ATT_STORE_G(o3, 3)
#undef ATT_STORE_G
}

// ---------------------------------------------------------------------------
extern "C" void kernel_launch(void* const* d_in, const int* in_sizes, int n_in,
                              void* d_out, int out_size, void* d_ws, size_t ws_size,
                              hipStream_t stream) {
    const float* Q = (const float*)d_in[0];
    const float* K = (const float*)d_in[1];
    const float* V = (const float*)d_in[2];
    float* O = (float*)d_out;

    const size_t tsz  = (size_t)ATT_B * ATT_N * ATT_D;   // elements per tensor
    const size_t need = 3 * tsz * sizeof(_Float16);      // Qh + Kh + Vth = 24 MB

    if (ws_size >= need) {
        _Float16* Qh  = (_Float16*)d_ws;
        _Float16* Kh  = Qh + tsz;
        _Float16* Vth = Kh + tsz;

        const int cvt_blocks = (int)(tsz / (8 * 256));
        cvt_f32_to_f16_kernel<<<cvt_blocks, 256, 0, stream>>>(Q, Qh, LOG2E);  // fold log2(e) into Q
        cvt_f32_to_f16_kernel<<<cvt_blocks, 256, 0, stream>>>(K, Kh, 1.0f);
        transpose_cvt_kernel<<<dim3(ATT_N / 64, ATT_B), 256, 0, stream>>>(V, Vth);

        attn_wmma_f16_kernel<<<dim3(ATT_N / 64, ATT_B), 128, 0, stream>>>(Qh, Kh, Vth, O);
    } else {
        attn_wmma_f32src_kernel<<<dim3(ATT_N / 16, ATT_B), 32, 0, stream>>>(Q, K, V, O);
    }
}
